// SpatialEncoder_80281528696953
// MI455X (gfx1250) — matrix operand order repaired and checked by reading, not compiled, not run on hardware
//
#include <hip/hip_runtime.h>
#include <hip/hip_bf16.h>
#include <cstddef>

typedef _Float16 v16h __attribute__((ext_vector_type(16)));
typedef float    v8f  __attribute__((ext_vector_type(8)));

#define NNODES 40000
#define NEDGES 640000
#define ETOT   (NNODES + NEDGES)

__device__ inline void atomicMaxF(float* addr, float val) {
  int* ai = reinterpret_cast<int*>(addr);
  int cur = __float_as_int(*addr);
  while (__int_as_float(cur) < val) {
    int prev = atomicCAS(ai, cur, __float_as_int(val));
    if (prev == cur) break;
    cur = prev;
  }
}

// ---------------------------------------------------------------------------
// Fragment loaders (contiguous k order, same in A and B, wave32):
//   A 16x32 f16: lanes 0-15 -> M=lane, K=[0..15]; lanes 16-31: K +16.
//   B 32x16 f16: lanes 0-15 -> N=lane, K=[0..15]; lanes 16-31: K +16.
// Rows are fp32 in memory; convert on the fly (v_cvt_pk_f16_f32).
// ---------------------------------------------------------------------------
__device__ inline v16h load_frag_a(const float* __restrict__ row, int k0, int hi) {
  const float4* p0 = (const float4*)(row + k0 + (hi << 4));
  const float4* p1 = (const float4*)(row + k0 + (hi << 4) + 8);
  float4 x0 = p0[0], x1 = p0[1];
  float4 y0 = p1[0], y1 = p1[1];
  v16h f;
  f[0]=(_Float16)x0.x;  f[1]=(_Float16)x0.y;  f[2]=(_Float16)x0.z;  f[3]=(_Float16)x0.w;
  f[4]=(_Float16)x1.x;  f[5]=(_Float16)x1.y;  f[6]=(_Float16)x1.z;  f[7]=(_Float16)x1.w;
  f[8]=(_Float16)y0.x;  f[9]=(_Float16)y0.y;  f[10]=(_Float16)y0.z; f[11]=(_Float16)y0.w;
  f[12]=(_Float16)y1.x; f[13]=(_Float16)y1.y; f[14]=(_Float16)y1.z; f[15]=(_Float16)y1.w;
  return f;
}

__device__ inline v16h load_frag_b(const float* __restrict__ row, int k0, int hi) {
  const float4* q = (const float4*)(row + k0 + (hi << 4));
  float4 w0=q[0], w1=q[1], w2=q[2], w3=q[3];
  v16h f;
  f[0]=(_Float16)w0.x;  f[1]=(_Float16)w0.y;  f[2]=(_Float16)w0.z;  f[3]=(_Float16)w0.w;
  f[4]=(_Float16)w1.x;  f[5]=(_Float16)w1.y;  f[6]=(_Float16)w1.z;  f[7]=(_Float16)w1.w;
  f[8]=(_Float16)w2.x;  f[9]=(_Float16)w2.y;  f[10]=(_Float16)w2.z; f[11]=(_Float16)w2.w;
  f[12]=(_Float16)w3.x; f[13]=(_Float16)w3.y; f[14]=(_Float16)w3.z; f[15]=(_Float16)w3.w;
  return f;
}

// ---------------------------------------------------------------------------
// WMMA GEMM:  C[n,M] = act( (A[n,K] @ W[M,K]^T + bias) * gamma*gscale + beta )
// One wave computes a 16x32 output tile: one A fragment feeds two WMMAs.
// K is compile-time -> k-loop fully unrolled, loads pipelined ahead of WMMAs.
// ---------------------------------------------------------------------------
template<int K, bool HAS_BIAS, bool HAS_BN, bool RELU>
__global__ __launch_bounds__(256) void k_gemm(
    const float* __restrict__ A, const float* __restrict__ W,
    const float* __restrict__ bias, const float* __restrict__ gamma,
    const float* __restrict__ beta, float gscale,
    float* __restrict__ C, int n, int M)
{
  const int wave = (int)((blockIdx.x * (unsigned)blockDim.x + threadIdx.x) >> 5);
  const int lane = threadIdx.x & 31;
  const int tiles_n = M >> 5;                 // 32-wide column tiles
  const int tiles = (n >> 4) * tiles_n;
  if (wave >= tiles) return;
  const int tm = (wave / tiles_n) << 4;
  const int tn = (wave % tiles_n) << 5;
  const int l15 = lane & 15;
  const int hi  = lane >> 4;

  const float* arow  = A + (size_t)(tm + l15) * K;
  const float* wrow0 = W + (size_t)(tn + l15) * K;
  const float* wrow1 = W + (size_t)(tn + 16 + l15) * K;

  v8f acc0 = {}, acc1 = {};
#pragma unroll
  for (int k0 = 0; k0 < K; k0 += 32) {
    v16h af  = load_frag_a(arow,  k0, hi);
    v16h bf0 = load_frag_b(wrow0, k0, hi);
    v16h bf1 = load_frag_b(wrow1, k0, hi);
    acc0 = __builtin_amdgcn_wmma_f32_16x16x32_f16(false, af, false, bf0, (short)0,
                                                  acc0, false, false);
    acc1 = __builtin_amdgcn_wmma_f32_16x16x32_f16(false, af, false, bf1, (short)0,
                                                  acc1, false, false);
  }

#pragma unroll
  for (int half = 0; half < 2; ++half) {
    const v8f& acc = half ? acc1 : acc0;
    const int col  = tn + (half << 4) + l15;
    float bi = 0.0f, ga = 1.0f, be = 0.0f;
    if constexpr (HAS_BIAS) bi = bias[col];
    if constexpr (HAS_BN)  { ga = gamma[col] * gscale; be = beta[col]; }
#pragma unroll
    for (int r = 0; r < 8; ++r) {
      int row = tm + (hi << 3) + r;
      float v = acc[r] + bi;
      if constexpr (HAS_BN) v = v * ga + be;
      if constexpr (RELU)   v = fmaxf(v, 0.0f);
      C[(size_t)row * M + col] = v;
    }
  }
}

// ---------------------------------------------------------------------------
// e_src[n,h] = <h[n,h,:], a_s[h,:]>, e_dst likewise
// ---------------------------------------------------------------------------
__global__ __launch_bounds__(256) void k_attn_scores(
    const float* __restrict__ h, const float* __restrict__ a_s,
    const float* __restrict__ a_d, float* __restrict__ esrc,
    float* __restrict__ edst, int n, int H, int C)
{
  int t = blockIdx.x * blockDim.x + threadIdx.x;
  if (t >= n * H) return;
  int node = t / H, hd = t % H;
  const float* hp = h + (size_t)node * H * C + (size_t)hd * C;
  const float* as = a_s + (size_t)hd * C;
  const float* ad = a_d + (size_t)hd * C;
  float s = 0.f, d = 0.f;
  for (int c = 0; c < C; c += 4) {
    float4 hv = *(const float4*)(hp + c);
    float4 av = *(const float4*)(as + c);
    float4 dv = *(const float4*)(ad + c);
    s += hv.x*av.x + hv.y*av.y + hv.z*av.z + hv.w*av.w;
    d += hv.x*dv.x + hv.y*dv.y + hv.z*dv.z + hv.w*dv.w;
  }
  esrc[t] = s; edst[t] = d;
}

__global__ __launch_bounds__(256) void k_fill(float* __restrict__ p, float v, int count) {
  int t = blockIdx.x * blockDim.x + threadIdx.x;
  if (t < count) p[t] = v;
}

__global__ __launch_bounds__(256) void k_init_bias(float* __restrict__ out,
                                                   const float* __restrict__ bias,
                                                   int n, int F) {
  int t = blockIdx.x * blockDim.x + threadIdx.x;
  if (t >= n * F) return;
  out[t] = bias[t % F];
}

__global__ __launch_bounds__(256) void k_edge_logits(
    const int* __restrict__ ei, const float* __restrict__ esrc,
    const float* __restrict__ edst, float* __restrict__ ebuf,
    float* __restrict__ emax, int E, int ET, int H)
{
  int t = blockIdx.x * blockDim.x + threadIdx.x;
  if (t >= ET * H) return;
  int e = t / H, hd = t % H;
  int s = (e < E) ? ei[e]     : (e - E);   // self loops appended
  int d = (e < E) ? ei[E + e] : (e - E);
  float v = esrc[s * H + hd] + edst[d * H + hd];
  v = (v >= 0.f) ? v : 0.2f * v;           // leaky_relu(0.2)
  ebuf[t] = v;
  atomicMaxF(&emax[d * H + hd], v);
}

__global__ __launch_bounds__(256) void k_edge_exp(
    const int* __restrict__ ei, float* __restrict__ ebuf,
    const float* __restrict__ emax, float* __restrict__ den,
    int E, int ET, int H)
{
  int t = blockIdx.x * blockDim.x + threadIdx.x;
  if (t >= ET * H) return;
  int e = t / H, hd = t % H;
  int d = (e < E) ? ei[E + e] : (e - E);
  float ex = __expf(ebuf[t] - emax[d * H + hd]);
  ebuf[t] = ex;
  atomicAdd(&den[d * H + hd], ex);
}

__global__ __launch_bounds__(256) void k_edge_alpha(
    const int* __restrict__ ei, float* __restrict__ ebuf,
    const float* __restrict__ den, int E, int ET, int H)
{
  int t = blockIdx.x * blockDim.x + threadIdx.x;
  if (t >= ET * H) return;
  int e = t / H, hd = t % H;
  int d = (e < E) ? ei[E + e] : (e - E);
  ebuf[t] = ebuf[t] / (den[d * H + hd] + 1e-16f);
}

// out[dst, f..f+3] += alpha[e, f/C] * h[src, f..f+3]; one float4 per thread
__global__ __launch_bounds__(256) void k_aggregate(
    const float* __restrict__ h, const float* __restrict__ alpha,
    const int* __restrict__ ei, float* __restrict__ out,
    int E, int ET, int H, int C)
{
  int F = H * C;
  int Q = F >> 2;                            // float4 groups per row
  int t = blockIdx.x * blockDim.x + threadIdx.x;
  if (t >= ET * Q) return;
  int e = t / Q, f = (t % Q) << 2;
  int hd = f / C;
  int s = (e < E) ? ei[e]     : (e - E);
  int d = (e < E) ? ei[E + e] : (e - E);
  float a = alpha[e * H + hd];
  float4 hv = *(const float4*)(h + (size_t)s * F + f);
  float* op = out + (size_t)d * F + f;
  atomicAdd(op + 0, a * hv.x);
  atomicAdd(op + 1, a * hv.y);
  atomicAdd(op + 2, a * hv.z);
  atomicAdd(op + 3, a * hv.w);
}

// wave-per-node LayerNorm + ReLU (wave32; F is 128 or 256)
__global__ __launch_bounds__(256) void k_ln_relu(
    float* __restrict__ x, const float* __restrict__ g,
    const float* __restrict__ b, int n, int F)
{
  int wave = (int)((blockIdx.x * (unsigned)blockDim.x + threadIdx.x) >> 5);
  int lane = threadIdx.x & 31;
  if (wave >= n) return;
  float* row = x + (size_t)wave * F;
  int per = F >> 5;
  float s = 0.f, q = 0.f;
  for (int i = 0; i < per; ++i) {
    float v = row[lane + (i << 5)];
    s += v; q += v * v;
  }
  for (int m = 16; m >= 1; m >>= 1) {
    s += __shfl_xor(s, m);
    q += __shfl_xor(q, m);
  }
  float mu  = s / F;
  float var = q / F - mu * mu;
  float inv = rsqrtf(var + 1e-5f);
  for (int i = 0; i < per; ++i) {
    int f = lane + (i << 5);
    float v = (row[f] - mu) * inv * g[f] + b[f];
    row[f] = fmaxf(v, 0.0f);
  }
}

// pointnet layer 1: pos[n,3] -> 64, fused eval-BN + ReLU
__global__ __launch_bounds__(256) void k_pn1(
    const float* __restrict__ pos, const float* __restrict__ pw1,
    const float* __restrict__ pb1, const float* __restrict__ g,
    const float* __restrict__ bb, float* __restrict__ out, int n)
{
  int t = blockIdx.x * blockDim.x + threadIdx.x;
  if (t >= n * 64) return;
  int node = t >> 6, o = t & 63;
  const float* pp = pos + (size_t)node * 3;
  const float* w  = pw1 + (size_t)o * 3;
  float acc = pp[0]*w[0] + pp[1]*w[1] + pp[2]*w[2] + pb1[o];
  float rs  = rsqrtf(1.0f + 1e-5f);
  out[t] = fmaxf(acc * g[o] * rs + bb[o], 0.0f);
}

// combined[n,256] = [x3[n,128] | pos_features[n,128]]  (float4 granularity)
__global__ __launch_bounds__(256) void k_pack(
    const float* __restrict__ x3, const float* __restrict__ pf,
    float* __restrict__ out, int n)
{
  int t = blockIdx.x * blockDim.x + threadIdx.x;
  if (t >= n * 64) return;                   // 64 float4s per node
  int node = t >> 6, g4 = t & 63;
  const float* srcp = (g4 < 32) ? (x3 + (size_t)node * 128 + (g4 << 2))
                                : (pf + (size_t)node * 128 + ((g4 - 32) << 2));
  *(float4*)(out + (size_t)node * 256 + (g4 << 2)) = *(const float4*)srcp;
}

// ---------------------------------------------------------------------------
extern "C" void kernel_launch(void* const* d_in, const int* in_sizes, int n_in,
                              void* d_out, int out_size, void* d_ws, size_t ws_size,
                              hipStream_t stream) {
  (void)in_sizes; (void)n_in; (void)out_size; (void)ws_size;
  constexpr int N_ = NNODES, E_ = NEDGES, ET_ = ETOT;

  const float* x     = (const float*)d_in[0];
  const int*   ei    = (const int*)  d_in[1];
  const float* pos   = (const float*)d_in[2];
  const float* W1    = (const float*)d_in[3];
  const float* a_s1  = (const float*)d_in[4];
  const float* a_d1  = (const float*)d_in[5];
  const float* bg1   = (const float*)d_in[6];
  const float* ln1g  = (const float*)d_in[7];
  const float* ln1b  = (const float*)d_in[8];
  const float* W2    = (const float*)d_in[9];
  const float* a_s2  = (const float*)d_in[10];
  const float* a_d2  = (const float*)d_in[11];
  const float* bg2   = (const float*)d_in[12];
  const float* ln2g  = (const float*)d_in[13];
  const float* ln2b  = (const float*)d_in[14];
  const float* W3    = (const float*)d_in[15];
  const float* a_s3  = (const float*)d_in[16];
  const float* a_d3  = (const float*)d_in[17];
  const float* bg3   = (const float*)d_in[18];
  const float* pw1   = (const float*)d_in[19];
  const float* pb1   = (const float*)d_in[20];
  const float* bn1g  = (const float*)d_in[21];
  const float* bn1b  = (const float*)d_in[22];
  const float* pw2   = (const float*)d_in[23];
  const float* pb2   = (const float*)d_in[24];
  const float* bn2g  = (const float*)d_in[25];
  const float* bn2b  = (const float*)d_in[26];
  const float* pw3   = (const float*)d_in[27];
  const float* pb3   = (const float*)d_in[28];
  const float* fw    = (const float*)d_in[29];
  const float* fb    = (const float*)d_in[30];

  float* out = (float*)d_out;                 // [output | x3 | pos_features]
  float* x3  = out + (size_t)N_ * 128;
  float* pf  = out + (size_t)2 * N_ * 128;

  float* ws   = (float*)d_ws;
  float* hA   = ws;                              // [N,256]  h of current layer
  float* hB   = hA   + (size_t)N_ * 256;         // [N,256]  aggregated / x_l
  float* ebuf = hB   + (size_t)N_ * 256;         // [ET,4]   logits -> exp -> alpha
  float* emax = ebuf + (size_t)ET_ * 4;          // [N,4]
  float* den  = emax + (size_t)N_ * 4;           // [N,4]
  float* esrc = den  + (size_t)N_ * 4;           // [N,4]
  float* edst = esrc + (size_t)N_ * 4;           // [N,4]

  const float rsq_bn = 0.9999950000374997f;      // rsqrt(1 + 1e-5)

  auto gblk = [&](int M) { return ((N_ / 16) * (M / 32) + 7) / 8; };

  // edge softmax + aggregation for one GAT layer (h already in hA)
  auto attn = [&](const float* a_s, const float* a_d, const float* bg,
                  int H, int C, float* aggOut) {
    int F = H * C;
    k_attn_scores<<<(N_ * H + 255) / 256, 256, 0, stream>>>(hA, a_s, a_d,
                                                            esrc, edst, N_, H, C);
    k_fill<<<(N_ * H + 255) / 256, 256, 0, stream>>>(emax, -1e30f, N_ * H);
    k_fill<<<(N_ * H + 255) / 256, 256, 0, stream>>>(den, 0.0f, N_ * H);
    k_init_bias<<<(N_ * F + 255) / 256, 256, 0, stream>>>(aggOut, bg, N_, F);
    int eh = ET_ * H;
    k_edge_logits<<<(eh + 255) / 256, 256, 0, stream>>>(ei, esrc, edst, ebuf,
                                                        emax, E_, ET_, H);
    k_edge_exp<<<(eh + 255) / 256, 256, 0, stream>>>(ei, ebuf, emax, den,
                                                     E_, ET_, H);
    k_edge_alpha<<<(eh + 255) / 256, 256, 0, stream>>>(ei, ebuf, den, E_, ET_, H);
    int tot = ET_ * (F >> 2);                 // float4 granularity, <= 43.5M
    k_aggregate<<<(tot + 255) / 256, 256, 0, stream>>>(hA, ebuf, ei, aggOut,
                                                       E_, ET_, H, C);
  };

  // ---- GAT layer 1: 64 -> 4x64, LN, ReLU
  k_gemm<64, false, false, false><<<gblk(256), 256, 0, stream>>>(
      x, W1, nullptr, nullptr, nullptr, 1.f, hA, N_, 256);
  attn(a_s1, a_d1, bg1, 4, 64, hB);
  k_ln_relu<<<(N_ + 7) / 8, 256, 0, stream>>>(hB, ln1g, ln1b, N_, 256);

  // ---- GAT layer 2: 256 -> 2x64, LN, ReLU
  k_gemm<256, false, false, false><<<gblk(128), 256, 0, stream>>>(
      hB, W2, nullptr, nullptr, nullptr, 1.f, hA, N_, 128);
  attn(a_s2, a_d2, bg2, 2, 64, hB);
  k_ln_relu<<<(N_ + 7) / 8, 256, 0, stream>>>(hB, ln2g, ln2b, N_, 128);

  // ---- GAT layer 3: 128 -> 1x128  -> x3 (second output)
  k_gemm<128, false, false, false><<<gblk(128), 256, 0, stream>>>(
      hB, W3, nullptr, nullptr, nullptr, 1.f, hA, N_, 128);
  attn(a_s3, a_d3, bg3, 1, 128, x3);

  // ---- PointNet (hB is free now): 3 -> 64 -> 128 -> 128
  float* p1 = hB;                      // [N,64]
  float* p2 = hB + (size_t)N_ * 64;    // [N,128]
  k_pn1<<<(N_ * 64 + 255) / 256, 256, 0, stream>>>(pos, pw1, pb1, bn1g, bn1b, p1, N_);
  k_gemm<64, true, true, true><<<gblk(128), 256, 0, stream>>>(
      p1, pw2, pb2, bn2g, bn2b, rsq_bn, p2, N_, 128);
  k_gemm<128, true, false, false><<<gblk(128), 256, 0, stream>>>(
      p2, pw3, pb3, nullptr, nullptr, 1.f, pf, N_, 128);

  // ---- fusion: concat(x3, pos_features) @ fw^T + fb -> output
  k_pack<<<(N_ * 64 + 255) / 256, 256, 0, stream>>>(x3, pf, hA, N_);
  k_gemm<256, true, false, false><<<gblk(128), 256, 0, stream>>>(
      hA, fw, fb, nullptr, nullptr, 1.f, out, N_, 128);
}
